// DeepMDsimpleForces_481036337907
// MI455X (gfx1250) — matrix operand order repaired and checked by reading, not compiled, not run on hardware
//
#include <hip/hip_runtime.h>
#include <hip/hip_bf16.h>
#include <math.h>

// ---------------- problem constants ----------------
#define BB 4
#define NN 4096
#define KK 32
#define NFREQ 512
#define CC 4
#define NA (BB*NN)            // 16384 atoms
#define LBOXF 128.0f
#define TWO_PI 6.2831853071795864f
#define INV_TWO_PI 0.15915494309189535f
#define KSTEP 0.04908738521234052f   // 2*pi/LBOX
#define TWO_LOG2E 2.8853900817779268f

typedef __attribute__((ext_vector_type(16))) _Float16 v16h;
typedef __attribute__((ext_vector_type(8)))  float    v8f;

struct Pyr { const float* W[5]; const float* b[5]; };

struct Params {
  const float* x; const int* neigh;
  Pyr pd, pi, plr, pfit;
  const float* lin_w; const float* lin_b; const float* mult;
  float *rho_re, *rho_im, *lrv, *D, *actf, *gD, *glr, *grho_re, *grho_im;
  float *energy, *forces;
};

// ---------------- small helpers ----------------
__device__ __forceinline__ void sincos_red(float ph, float* s, float* c) {
  float r = ph - TWO_PI * rintf(ph * INV_TWO_PI);
  __sincosf(r, s, c);
}
__device__ __forceinline__ float kgridf(int f) { return KSTEP * (float)(f - NFREQ/2); }

// Branchless tanh: one v_exp_f32 + v_rcp_f32, no divergence, saturates to +-1.
__device__ __forceinline__ float fast_tanh(float x) {
  float ax = fabsf(x);
  float e2 = __builtin_amdgcn_exp2f(ax * TWO_LOG2E);     // exp(2|x|)
  float t  = 1.0f - 2.0f * __builtin_amdgcn_rcpf(e2 + 1.0f);
  return copysignf(t, x);
}

// ---------------- WMMA wave32 fragment helpers (A/B: contiguous k order used here; C/D: ISA 7.12.2) ----
// A (16x32 f16) from LDS, unconditional: lane L: m=L&15, half=L>>4;
// elem i -> K = 16*half + i  (one contiguous 32B run per lane)
__device__ __forceinline__ v16h a_frag(const _Float16* A, int lda) {
  const int lane = threadIdx.x & 31;
  const int m = lane & 15, half = lane >> 4;
  v16h r;
#pragma unroll
  for (int i = 0; i < 16; ++i) {
    int k = 16*half + i;
    r[i] = A[m*lda + k];
  }
  return r;
}
// B (32x16 f16), full K=32 valid: B[k][n] = W[(koff+k)*ldw + noff + n]
__device__ __forceinline__ v16h b_frag_full(const float* W, int ldw, int koff, int noff) {
  const int lane = threadIdx.x & 31;
  const int n = lane & 15, half = lane >> 4;
  v16h r;
#pragma unroll
  for (int i = 0; i < 16; ++i) {
    int k = 16*half + i;
    r[i] = (_Float16)W[(koff + k)*ldw + noff + n];
  }
  return r;
}
// B for a 16x32 weight (K=16 real, upper 16 zero). Branchless: clamped address
// + value select (half==1 lanes are all-zero).
__device__ __forceinline__ v16h b_frag_w4(const float* W, int noff) {
  const int lane = threadIdx.x & 31;
  const int n = lane & 15, half = lane >> 4;
  v16h r;
#pragma unroll
  for (int i = 0; i < 16; ++i) {
    int k = 16*half + i;
    float v = W[(k & 15)*32 + noff + n];         // always in-bounds
    r[i] = (_Float16)((half == 0) ? v : 0.0f);   // v_cndmask, no branch
  }
  return r;
}
// Transposed weight fragment, K=16 real: B[k][n] = W[(noff+n)*ldw + k]
__device__ __forceinline__ v16h b_frag_T16(const float* W, int ldw, int noff) {
  const int lane = threadIdx.x & 31;
  const int n = lane & 15, half = lane >> 4;
  v16h r;
#pragma unroll
  for (int i = 0; i < 16; ++i) {
    int k = 16*half + i;
    float v = W[(noff + n)*ldw + (k & 15)];
    r[i] = (_Float16)((half == 0) ? v : 0.0f);
  }
  return r;
}
__device__ __forceinline__ v8f wmma_f16(v16h a, v16h b, v8f c) {
  return __builtin_amdgcn_wmma_f32_16x16x32_f16(false, a, false, b, (short)0, c, false, false);
}
// C/D (16x16 f32): lane L: n=L&15, half=L>>4; c[r] = C[8*half + r][n]
__device__ __forceinline__ void c_store_lds(float* Cb, int ldc, v8f c) {
  const int lane = threadIdx.x & 31;
  const int n = lane & 15, half = lane >> 4;
#pragma unroll
  for (int r = 0; r < 8; ++r) Cb[(8*half + r)*ldc + n] = c[r];
}

// ---------------- scalar pyramid heads ----------------
__device__ __forceinline__ void pyr_head_tanh(float u, const Pyr& P, float x3[16]) {
  float x0[2], x1[4], x2[8];
#pragma unroll
  for (int j = 0; j < 2; ++j) x0[j] = fast_tanh(u * P.W[0][j] + P.b[0][j]);
#pragma unroll
  for (int j = 0; j < 4; ++j) {
    float p = P.b[1][j];
#pragma unroll
    for (int k = 0; k < 2; ++k) p += x0[k] * P.W[1][k*4 + j];
    x1[j] = x0[j & 1] + fast_tanh(p);
  }
#pragma unroll
  for (int j = 0; j < 8; ++j) {
    float p = P.b[2][j];
#pragma unroll
    for (int k = 0; k < 4; ++k) p += x1[k] * P.W[2][k*8 + j];
    x2[j] = x1[j & 3] + fast_tanh(p);
  }
#pragma unroll
  for (int j = 0; j < 16; ++j) {
    float p = P.b[3][j];
#pragma unroll
    for (int k = 0; k < 8; ++k) p += x2[k] * P.W[3][k*16 + j];
    x3[j] = x2[j & 7] + fast_tanh(p);
  }
}

// forward-mode (JVP) version: tangent wrt scalar input u
__device__ __forceinline__ void pyr_head_tanh_jvp(float u, const Pyr& P,
                                                  float x3[16], float t3[16]) {
  float x0[2], t0[2], x1[4], t1[4], x2[8], t2[8];
#pragma unroll
  for (int j = 0; j < 2; ++j) {
    float y = fast_tanh(u * P.W[0][j] + P.b[0][j]);
    x0[j] = y; t0[j] = (1.0f - y*y) * P.W[0][j];
  }
#pragma unroll
  for (int j = 0; j < 4; ++j) {
    float p = P.b[1][j], tp = 0.0f;
#pragma unroll
    for (int k = 0; k < 2; ++k) { p += x0[k]*P.W[1][k*4+j]; tp += t0[k]*P.W[1][k*4+j]; }
    float y = fast_tanh(p);
    x1[j] = x0[j & 1] + y; t1[j] = t0[j & 1] + (1.0f - y*y)*tp;
  }
#pragma unroll
  for (int j = 0; j < 8; ++j) {
    float p = P.b[2][j], tp = 0.0f;
#pragma unroll
    for (int k = 0; k < 4; ++k) { p += x1[k]*P.W[2][k*8+j]; tp += t1[k]*P.W[2][k*8+j]; }
    float y = fast_tanh(p);
    x2[j] = x1[j & 3] + y; t2[j] = t1[j & 3] + (1.0f - y*y)*tp;
  }
#pragma unroll
  for (int j = 0; j < 16; ++j) {
    float p = P.b[3][j], tp = 0.0f;
#pragma unroll
    for (int k = 0; k < 8; ++k) { p += x2[k]*P.W[3][k*16+j]; tp += t2[k]*P.W[3][k*16+j]; }
    float y = fast_tanh(p);
    x3[j] = x2[j & 7] + y; t3[j] = t2[j & 7] + (1.0f - y*y)*tp;
  }
}

// pyr_lr forward (relu): 4->2 plain, then doubling residual up to 32
__device__ __forceinline__ void pyr_lr_fwd(const float in4[4], const Pyr& P, float out[32]) {
  float x0[2], x1[4], x2[8], x3[16];
#pragma unroll
  for (int j = 0; j < 2; ++j) {
    float p = P.b[0][j];
#pragma unroll
    for (int k = 0; k < 4; ++k) p += in4[k]*P.W[0][k*2+j];
    x0[j] = fmaxf(p, 0.0f);
  }
#pragma unroll
  for (int j = 0; j < 4; ++j) {
    float p = P.b[1][j];
#pragma unroll
    for (int k = 0; k < 2; ++k) p += x0[k]*P.W[1][k*4+j];
    x1[j] = x0[j & 1] + fmaxf(p, 0.0f);
  }
#pragma unroll
  for (int j = 0; j < 8; ++j) {
    float p = P.b[2][j];
#pragma unroll
    for (int k = 0; k < 4; ++k) p += x1[k]*P.W[2][k*8+j];
    x2[j] = x1[j & 3] + fmaxf(p, 0.0f);
  }
#pragma unroll
  for (int j = 0; j < 16; ++j) {
    float p = P.b[3][j];
#pragma unroll
    for (int k = 0; k < 8; ++k) p += x2[k]*P.W[3][k*16+j];
    x3[j] = x2[j & 7] + fmaxf(p, 0.0f);
  }
#pragma unroll
  for (int j = 0; j < 32; ++j) {
    float p = P.b[4][j];
#pragma unroll
    for (int k = 0; k < 16; ++k) p += x3[k]*P.W[4][k*32+j];
    out[j] = x3[j & 15] + fmaxf(p, 0.0f);
  }
}

// pyr_lr backward: recompute forward, VJP of g4 -> gin (4)
__device__ __forceinline__ void pyr_lr_bwd(const float in4[4], const Pyr& P,
                                           const float g4[32], float gin[4]) {
  float x0[2], x1[4], x2[8], x3[16];
  float y1[4], y2[8], y3[16], y4[32];
#pragma unroll
  for (int j = 0; j < 2; ++j) {
    float p = P.b[0][j];
#pragma unroll
    for (int k = 0; k < 4; ++k) p += in4[k]*P.W[0][k*2+j];
    x0[j] = fmaxf(p, 0.0f);
  }
#pragma unroll
  for (int j = 0; j < 4; ++j) {
    float p = P.b[1][j];
#pragma unroll
    for (int k = 0; k < 2; ++k) p += x0[k]*P.W[1][k*4+j];
    y1[j] = fmaxf(p, 0.0f); x1[j] = x0[j & 1] + y1[j];
  }
#pragma unroll
  for (int j = 0; j < 8; ++j) {
    float p = P.b[2][j];
#pragma unroll
    for (int k = 0; k < 4; ++k) p += x1[k]*P.W[2][k*8+j];
    y2[j] = fmaxf(p, 0.0f); x2[j] = x1[j & 3] + y2[j];
  }
#pragma unroll
  for (int j = 0; j < 16; ++j) {
    float p = P.b[3][j];
#pragma unroll
    for (int k = 0; k < 8; ++k) p += x2[k]*P.W[3][k*16+j];
    y3[j] = fmaxf(p, 0.0f); x3[j] = x2[j & 7] + y3[j];
  }
#pragma unroll
  for (int j = 0; j < 32; ++j) {
    float p = P.b[4][j];
#pragma unroll
    for (int k = 0; k < 16; ++k) p += x3[k]*P.W[4][k*32+j];
    y4[j] = fmaxf(p, 0.0f);
  }
  float g3[16];
#pragma unroll
  for (int k = 0; k < 16; ++k) {
    float acc = g4[k] + g4[16 + k];
#pragma unroll
    for (int j = 0; j < 32; ++j) acc += (y4[j] > 0.0f ? g4[j] : 0.0f) * P.W[4][k*32+j];
    g3[k] = acc;
  }
  float g2[8];
#pragma unroll
  for (int k = 0; k < 8; ++k) {
    float acc = g3[k] + g3[8 + k];
#pragma unroll
    for (int j = 0; j < 16; ++j) acc += (y3[j] > 0.0f ? g3[j] : 0.0f) * P.W[3][k*16+j];
    g2[k] = acc;
  }
  float g1[4];
#pragma unroll
  for (int k = 0; k < 4; ++k) {
    float acc = g2[k] + g2[4 + k];
#pragma unroll
    for (int j = 0; j < 8; ++j) acc += (y2[j] > 0.0f ? g2[j] : 0.0f) * P.W[2][k*8+j];
    g1[k] = acc;
  }
  float g0[2];
#pragma unroll
  for (int k = 0; k < 2; ++k) {
    float acc = g1[k] + g1[2 + k];
#pragma unroll
    for (int j = 0; j < 4; ++j) acc += (y1[j] > 0.0f ? g1[j] : 0.0f) * P.W[1][k*4+j];
    g0[k] = acc;
  }
#pragma unroll
  for (int k = 0; k < 4; ++k) {
    float acc = 0.0f;
#pragma unroll
    for (int j = 0; j < 2; ++j) acc += (x0[j] > 0.0f ? g0[j] : 0.0f) * P.W[0][k*2+j];
    gin[k] = acc;
  }
}

// ---------------- per-pyramid WMMA bodies (called twice, fully unrolled) -----
__device__ __forceinline__ void edge_pyr_fwd_body(
    const Pyr& PY, float u, float s,
    _Float16 (*st)[32], float (*bf)[32],
    v16h b0, v16h b1, int lane, float* Dout)
{
  float x3[16];
  pyr_head_tanh(u, PY, x3);
  __syncthreads();
#pragma unroll
  for (int k = 0; k < 16; ++k) st[lane][k] = (_Float16)x3[k];
#pragma unroll
  for (int k = 16; k < 32; ++k) st[lane][k] = (_Float16)0.0f;
  __syncthreads();
#pragma unroll
  for (int sub = 0; sub < 2; ++sub) {
    v16h af = a_frag(&st[sub*16][0], 32);
    { v8f c = {}; c = wmma_f16(af, b0, c); c_store_lds(&bf[sub*16][0], 32, c); }
    { v8f c = {}; c = wmma_f16(af, b1, c); c_store_lds(&bf[sub*16][16], 32, c); }
  }
  __syncthreads();
#pragma unroll
  for (int q = 0; q < 32; ++q) {
    float x4 = x3[q & 15] + fast_tanh(bf[lane][q] + PY.b[4][q]);
    bf[lane][q] = x4 * s;
  }
  __syncthreads();
  float colsum = 0.0f;
#pragma unroll
  for (int ee = 0; ee < 32; ++ee) colsum += bf[ee][lane];
  Dout[lane] = colsum;
  __syncthreads();
}

__device__ __forceinline__ void edge_pyr_bwd_body(
    const Pyr& PY, float u, bool isDist, float s,
    _Float16 (*sx)[32], _Float16 (*stg)[32], float (*bp)[32], float (*bt)[32],
    v16h b0, v16h b1, const float* gDbase, int lane,
    float& gf1, float& gs)
{
  float x3[16], t3[16];
  pyr_head_tanh_jvp(u, PY, x3, t3);
  __syncthreads();
#pragma unroll
  for (int k = 0; k < 16; ++k) { sx[lane][k] = (_Float16)x3[k]; stg[lane][k] = (_Float16)t3[k]; }
#pragma unroll
  for (int k = 16; k < 32; ++k) { sx[lane][k] = (_Float16)0.0f; stg[lane][k] = (_Float16)0.0f; }
  __syncthreads();
#pragma unroll
  for (int sub = 0; sub < 2; ++sub) {
    v16h ax = a_frag(&sx[sub*16][0], 32);
    v16h at = a_frag(&stg[sub*16][0], 32);
    { v8f c = {}; c = wmma_f16(ax, b0, c); c_store_lds(&bp[sub*16][0], 32, c); }
    { v8f c = {}; c = wmma_f16(ax, b1, c); c_store_lds(&bp[sub*16][16], 32, c); }
    { v8f c = {}; c = wmma_f16(at, b0, c); c_store_lds(&bt[sub*16][0], 32, c); }
    { v8f c = {}; c = wmma_f16(at, b1, c); c_store_lds(&bt[sub*16][16], 32, c); }
  }
  __syncthreads();
#pragma unroll
  for (int q = 0; q < 32; ++q) {
    float th = fast_tanh(bp[lane][q] + PY.b[4][q]);
    float x4 = x3[q & 15] + th;
    float t4 = t3[q & 15] + (1.0f - th*th) * bt[lane][q];
    float g = gDbase[q];
    if (isDist) { gf1 += g * s * t4; gs += g * x4; }
    else        { gs  += g * (s * t4 + x4); }
  }
  __syncthreads();
}

// ---------------- kernels ----------------
__global__ __launch_bounds__(256) void k_init(Params P) {
  int i = blockIdx.x * 256 + threadIdx.x;
  if (i < BB + NA) { if (i < BB) P.energy[i] = 0.0f; else P.forces[i - BB] = 0.0f; }
  if (i < 2*BB*NFREQ) { if (i < BB*NFREQ) P.rho_re[i] = 0.0f; else P.rho_im[i - BB*NFREQ] = 0.0f; }
}

// rho[b,f] = sum_n exp(-i x_n k_f); split atoms over 8 chunks, atomic combine
__global__ __launch_bounds__(256) void k_rho(Params P) {
  const int id = blockIdx.x;
  const int b = id >> 4;
  const int rem = id & 15;
  const int f = ((rem & 1) << 8) + threadIdx.x;
  const int n0 = (rem >> 1) * (NN/8);
  const float kf = kgridf(f);
  const float* xb = P.x + b*NN;
  float ar = 0.0f, ai = 0.0f;
  for (int n = n0; n < n0 + NN/8; ++n) {
    float sn, cs; sincos_red(xb[n]*kf, &sn, &cs);
    ar += cs; ai -= sn;
  }
  atomicAdd(&P.rho_re[b*NFREQ + f], ar);
  atomicAdd(&P.rho_im[b*NFREQ + f], ai);
}

// lr[a,c] = (1/NF) sum_f mult[c,f]*(rho_re*cos(x k) - rho_im*sin(x k)); wave/atom
__global__ __launch_bounds__(128) void k_lr(Params P) {
  const int wave = threadIdx.x >> 5, lane = threadIdx.x & 31;
  const int a = blockIdx.x*4 + wave;
  const int b = a >> 12;
  const float xn = P.x[a];
  float acc[CC] = {0.f, 0.f, 0.f, 0.f};
  for (int it = 0; it < NFREQ/32; ++it) {
    int f = lane + (it << 5);
    float kf = kgridf(f);
    float sn, cs; sincos_red(xn*kf, &sn, &cs);
    float rr = P.rho_re[b*NFREQ + f], ri = P.rho_im[b*NFREQ + f];
    float e = rr*cs - ri*sn;
#pragma unroll
    for (int c = 0; c < CC; ++c) acc[c] += P.mult[c*NFREQ + f] * e;
  }
#pragma unroll
  for (int c = 0; c < CC; ++c) {
    float v = acc[c];
#pragma unroll
    for (int m = 16; m; m >>= 1) v += __shfl_xor(v, m, 32);
    if (lane == 0) P.lrv[a*CC + c] = v * (1.0f/NFREQ);
  }
}

// edge forward: wave = atom, lane = neighbor slot; WMMA for the 16->32 layer
__global__ __launch_bounds__(128) void k_edge_fwd(Params P) {
  __shared__ _Float16 stage[4][32][32];
  __shared__ float buf[4][32][32];
  const int wave = threadIdx.x >> 5, lane = threadIdx.x & 31;
  const int a = blockIdx.x*4 + wave;
  const int b = a >> 12;
  const int e = a*KK + lane;
  const int jn = P.neigh[e];
  const float xn = P.x[a];
  const float xj = P.x[b*NN + (jn < 0 ? 0 : jn)];
  float d = xj - xn;
  d -= LBOXF * rintf(d * (1.0f/LBOXF));
  const float r = fabsf(d);
  const bool valid = (jn >= 0) && (r > 1e-6f);
  const float rs = valid ? r : 1.0f;
  const float f0 = valid ? 1.0f/rs : 0.0f;
  const float f1 = valid ? d/(rs*rs) : 0.0f;
  const float s = f0;

  // loop-invariant weight fragments (branchless, built once)
  v16h bd0 = b_frag_w4(P.pd.W[4], 0);
  v16h bd1 = b_frag_w4(P.pd.W[4], 16);
  v16h bi0 = b_frag_w4(P.pi.W[4], 0);
  v16h bi1 = b_frag_w4(P.pi.W[4], 16);

  edge_pyr_fwd_body(P.pd, f1, s, stage[wave], buf[wave], bd0, bd1, lane,
                    P.D + (size_t)a*64);
  edge_pyr_fwd_body(P.pi, f0, s, stage[wave], buf[wave], bi0, bi1, lane,
                    P.D + (size_t)a*64 + 32);
}

// fit forward: wave = 32 atoms, WMMA 96->16 layer, scalar tail, store activations
__global__ __launch_bounds__(128) void k_fit_fwd(Params P) {
  __shared__ _Float16 inS[4][32][96];
  __shared__ float pbuf[4][32][16];
  const int wave = threadIdx.x >> 5, lane = threadIdx.x & 31;
  const int aBase = blockIdx.x*128 + wave*32;
  const int a = aBase + lane;
  const int b = a >> 12;
  _Float16 (*st)[96] = inS[wave];
#pragma unroll
  for (int i = 0; i < 64; ++i) st[lane][i] = (_Float16)P.D[(size_t)a*64 + i];
  float lr4[4];
#pragma unroll
  for (int c = 0; c < 4; ++c) lr4[c] = P.lrv[a*4 + c];
  float L3[32];
  pyr_lr_fwd(lr4, P.plr, L3);
#pragma unroll
  for (int i = 0; i < 32; ++i) st[lane][64 + i] = (_Float16)L3[i];
  __syncthreads();
  // loop-invariant weight fragments for the three K-chunks of W0 (96x16)
  v16h w0 = b_frag_full(P.pfit.W[0], 16, 0, 0);
  v16h w1 = b_frag_full(P.pfit.W[0], 16, 32, 0);
  v16h w2 = b_frag_full(P.pfit.W[0], 16, 64, 0);
  float (*pb)[16] = pbuf[wave];
#pragma unroll
  for (int sub = 0; sub < 2; ++sub) {
    v8f c = {};
    c = wmma_f16(a_frag(&st[sub*16][0], 96), w0, c);
    c = wmma_f16(a_frag(&st[sub*16][32], 96), w1, c);
    c = wmma_f16(a_frag(&st[sub*16][64], 96), w2, c);
    c_store_lds(&pb[sub*16][0], 16, c);
  }
  __syncthreads();
  float* act = P.actf + (size_t)a * 31;
  float x0[16];
#pragma unroll
  for (int i = 0; i < 16; ++i) { x0[i] = fast_tanh(pb[lane][i] + P.pfit.b[0][i]); act[i] = x0[i]; }
  float x1[8];
#pragma unroll
  for (int j = 0; j < 8; ++j) {
    float p = P.pfit.b[1][j];
#pragma unroll
    for (int k = 0; k < 16; ++k) p += x0[k]*P.pfit.W[1][k*8+j];
    x1[j] = fast_tanh(p); act[16 + j] = x1[j];
  }
  float x2[4];
#pragma unroll
  for (int j = 0; j < 4; ++j) {
    float p = P.pfit.b[2][j];
#pragma unroll
    for (int k = 0; k < 8; ++k) p += x1[k]*P.pfit.W[2][k*4+j];
    x2[j] = fast_tanh(p); act[24 + j] = x2[j];
  }
  float x3[2];
#pragma unroll
  for (int j = 0; j < 2; ++j) {
    float p = P.pfit.b[3][j];
#pragma unroll
    for (int k = 0; k < 4; ++k) p += x2[k]*P.pfit.W[3][k*2+j];
    x3[j] = fast_tanh(p); act[28 + j] = x3[j];
  }
  float p4 = P.pfit.b[4][0] + x3[0]*P.pfit.W[4][0] + x3[1]*P.pfit.W[4][1];
  float x4 = fast_tanh(p4); act[30] = x4;
  float F = x4 * P.lin_w[0] + P.lin_b[0];
  atomicAdd(&P.energy[b], F);
}

// fit backward: scalar tail backward, WMMA dX = dpre0 @ W0^T, then pyr_lr VJP
__global__ __launch_bounds__(128) void k_fit_bwd(Params P) {
  __shared__ _Float16 dst[4][32][32];
  __shared__ float g3b[4][32][32];
  const int wave = threadIdx.x >> 5, lane = threadIdx.x & 31;
  const int aBase = blockIdx.x*128 + wave*32;
  const int a = aBase + lane;
  const float* act = P.actf + (size_t)a * 31;
  float x0[16], x1[8], x2[4], x3[2], x4;
#pragma unroll
  for (int i = 0; i < 16; ++i) x0[i] = act[i];
#pragma unroll
  for (int i = 0; i < 8; ++i) x1[i] = act[16 + i];
#pragma unroll
  for (int i = 0; i < 4; ++i) x2[i] = act[24 + i];
#pragma unroll
  for (int i = 0; i < 2; ++i) x3[i] = act[28 + i];
  x4 = act[30];
  float dp4 = P.lin_w[0] * (1.0f - x4*x4);
  float dp3[2];
#pragma unroll
  for (int k = 0; k < 2; ++k) dp3[k] = dp4 * P.pfit.W[4][k] * (1.0f - x3[k]*x3[k]);
  float dp2[4];
#pragma unroll
  for (int k = 0; k < 4; ++k) {
    float acc = 0.0f;
#pragma unroll
    for (int j = 0; j < 2; ++j) acc += dp3[j]*P.pfit.W[3][k*2+j];
    dp2[k] = acc * (1.0f - x2[k]*x2[k]);
  }
  float dp1[8];
#pragma unroll
  for (int k = 0; k < 8; ++k) {
    float acc = 0.0f;
#pragma unroll
    for (int j = 0; j < 4; ++j) acc += dp2[j]*P.pfit.W[2][k*4+j];
    dp1[k] = acc * (1.0f - x1[k]*x1[k]);
  }
  float dp0[16];
#pragma unroll
  for (int k = 0; k < 16; ++k) {
    float acc = 0.0f;
#pragma unroll
    for (int j = 0; j < 8; ++j) acc += dp1[j]*P.pfit.W[1][k*8+j];
    dp0[k] = acc * (1.0f - x0[k]*x0[k]);
  }
  _Float16 (*st)[32] = dst[wave];
#pragma unroll
  for (int k = 0; k < 16; ++k) st[lane][k] = (_Float16)dp0[k];
#pragma unroll
  for (int k = 16; k < 32; ++k) st[lane][k] = (_Float16)0.0f;
  __syncthreads();
  // 6 loop-invariant transposed weight fragments for W0^T (16x96)
  v16h wt[6];
#pragma unroll
  for (int nc = 0; nc < 6; ++nc) wt[nc] = b_frag_T16(P.pfit.W[0], 16, nc*16);
  float (*gb)[32] = g3b[wave];
  const int nloc = lane & 15, half = lane >> 4;
#pragma unroll
  for (int sub = 0; sub < 2; ++sub) {
    v16h af = a_frag(&st[sub*16][0], 32);
#pragma unroll
    for (int nc = 0; nc < 6; ++nc) {
      v8f c = {};
      c = wmma_f16(af, wt[nc], c);
      if (nc < 4) {
#pragma unroll
        for (int r = 0; r < 8; ++r)
          P.gD[(size_t)(aBase + sub*16 + 8*half + r)*64 + nc*16 + nloc] = c[r];
      } else {
#pragma unroll
        for (int r = 0; r < 8; ++r)
          gb[sub*16 + 8*half + r][(nc - 4)*16 + nloc] = c[r];
      }
    }
  }
  __syncthreads();
  float gL3[32];
#pragma unroll
  for (int i = 0; i < 32; ++i) gL3[i] = gb[lane][i];
  float lr4[4];
#pragma unroll
  for (int c = 0; c < 4; ++c) lr4[c] = P.lrv[a*4 + c];
  float glr4[4];
  pyr_lr_bwd(lr4, P.plr, gL3, glr4);
#pragma unroll
  for (int c = 0; c < 4; ++c) P.glr[a*4 + c] = glr4[c];
}

// dE/drho adjoints: thread per (b,f), loop atoms
__global__ __launch_bounds__(256) void k_grho(Params P) {
  const int id = blockIdx.x*256 + threadIdx.x;
  const int b = id >> 9;
  const int f = id & (NFREQ - 1);
  const float kf = kgridf(f);
  const float m0 = P.mult[0*NFREQ+f], m1 = P.mult[1*NFREQ+f];
  const float m2 = P.mult[2*NFREQ+f], m3 = P.mult[3*NFREQ+f];
  const float* xb = P.x + b*NN;
  const float4* g4 = (const float4*)(P.glr + (size_t)b*NN*4);
  float ar = 0.0f, ai = 0.0f;
  for (int n = 0; n < NN; ++n) {
    float4 g = g4[n];
    float w = g.x*m0 + g.y*m1 + g.z*m2 + g.w*m3;
    float sn, cs; sincos_red(xb[n]*kf, &sn, &cs);
    ar += w*cs; ai -= w*sn;
  }
  P.grho_re[id] = ar * (1.0f/NFREQ);
  P.grho_im[id] = ai * (1.0f/NFREQ);
}

// NUFFT force: direct (own-phase) + indirect (through rho) terms; wave per atom
__global__ __launch_bounds__(128) void k_nufft_force(Params P) {
  const int wave = threadIdx.x >> 5, lane = threadIdx.x & 31;
  const int a = blockIdx.x*4 + wave;
  const int b = a >> 12;
  const float xn = P.x[a];
  const float g0 = P.glr[a*4+0], g1 = P.glr[a*4+1], g2 = P.glr[a*4+2], g3 = P.glr[a*4+3];
  float acc = 0.0f;
  for (int it = 0; it < NFREQ/32; ++it) {
    int f = lane + (it << 5);
    float kf = kgridf(f);
    float sn, cs; sincos_red(xn*kf, &sn, &cs);
    float mc = (g0*P.mult[0*NFREQ+f] + g1*P.mult[1*NFREQ+f] +
                g2*P.mult[2*NFREQ+f] + g3*P.mult[3*NFREQ+f]) * (1.0f/NFREQ);
    float rr = P.rho_re[b*NFREQ+f], ri = P.rho_im[b*NFREQ+f];
    float gr = P.grho_re[b*NFREQ+f], gi = P.grho_im[b*NFREQ+f];
    acc += kf * (mc*(-rr*sn - ri*cs) + (-gr*sn - gi*cs));
  }
#pragma unroll
  for (int m = 16; m; m >>= 1) acc += __shfl_xor(acc, m, 32);
  if (lane == 0) atomicAdd(&P.forces[a], -acc);
}

// edge backward: forward-mode tangents wrt f1/f0; WMMA value+tangent 16->32 layer
__global__ __launch_bounds__(128) void k_edge_bwd(Params P) {
  __shared__ _Float16 stX[4][32][32];
  __shared__ _Float16 stT[4][32][32];
  __shared__ float bufP[4][32][32];
  __shared__ float bufT[4][32][32];
  const int wave = threadIdx.x >> 5, lane = threadIdx.x & 31;
  const int a = blockIdx.x*4 + wave;
  const int b = a >> 12;
  const int e = a*KK + lane;
  const int jn = P.neigh[e];
  const float xn = P.x[a];
  const float xj = P.x[b*NN + (jn < 0 ? 0 : jn)];
  float d = xj - xn;
  d -= LBOXF * rintf(d * (1.0f/LBOXF));
  const float r = fabsf(d);
  const bool valid = (jn >= 0) && (r > 1e-6f);
  const float rs = valid ? r : 1.0f;
  const float f0 = valid ? 1.0f/rs : 0.0f;
  const float f1 = valid ? d/(rs*rs) : 0.0f;
  const float s = f0;

  v16h bd0 = b_frag_w4(P.pd.W[4], 0);
  v16h bd1 = b_frag_w4(P.pd.W[4], 16);
  v16h bi0 = b_frag_w4(P.pi.W[4], 0);
  v16h bi1 = b_frag_w4(P.pi.W[4], 16);

  float gf1 = 0.0f, gs = 0.0f;
  edge_pyr_bwd_body(P.pd, f1, true,  s, stX[wave], stT[wave], bufP[wave], bufT[wave],
                    bd0, bd1, P.gD + (size_t)a*64,      lane, gf1, gs);
  edge_pyr_bwd_body(P.pi, f0, false, s, stX[wave], stT[wave], bufP[wave], bufT[wave],
                    bi0, bi1, P.gD + (size_t)a*64 + 32, lane, gf1, gs);

  float gd = 0.0f;
  if (valid) {
    float inv_d2 = 1.0f / (d*d);
    float sgn = (d >= 0.0f) ? 1.0f : -1.0f;
    gd = gf1 * (-inv_d2) + gs * (-sgn * inv_d2);
  }
  float v = gd;
#pragma unroll
  for (int m = 16; m; m >>= 1) v += __shfl_xor(v, m, 32);
  if (lane == 0) atomicAdd(&P.forces[a], v);                 // force_n = -dE/dx_n
  if (valid) atomicAdd(&P.forces[b*NN + jn], -gd);           // force_j = -dE/dx_j
}

// ---------------- host launch ----------------
extern "C" void kernel_launch(void* const* d_in, const int* in_sizes, int n_in,
                              void* d_out, int out_size, void* d_ws, size_t ws_size,
                              hipStream_t stream) {
  (void)in_sizes; (void)n_in; (void)out_size; (void)ws_size;
  Params P;
  P.x = (const float*)d_in[0];
  P.neigh = (const int*)d_in[1];
  auto setPyr = [&](Pyr& Q, int base) {
    for (int l = 0; l < 5; ++l) {
      Q.W[l] = (const float*)d_in[base + 2*l];
      Q.b[l] = (const float*)d_in[base + 2*l + 1];
    }
  };
  setPyr(P.pd, 2); setPyr(P.pi, 12); setPyr(P.plr, 22); setPyr(P.pfit, 32);
  P.lin_w = (const float*)d_in[42];
  P.lin_b = (const float*)d_in[43];
  P.mult  = (const float*)d_in[44];

  float* ws = (float*)d_ws;
  size_t o = 0;
  P.rho_re  = ws + o; o += (size_t)BB*NFREQ;
  P.rho_im  = ws + o; o += (size_t)BB*NFREQ;
  P.lrv     = ws + o; o += (size_t)NA*CC;
  P.D       = ws + o; o += (size_t)NA*64;
  P.actf    = ws + o; o += (size_t)NA*31;
  P.gD      = ws + o; o += (size_t)NA*64;
  P.glr     = ws + o; o += (size_t)NA*CC;
  P.grho_re = ws + o; o += (size_t)BB*NFREQ;
  P.grho_im = ws + o; o += (size_t)BB*NFREQ;
  P.energy  = (float*)d_out;
  P.forces  = (float*)d_out + BB;

  k_init       <<<65,        256, 0, stream>>>(P);
  k_rho        <<<BB*16,     256, 0, stream>>>(P);
  k_lr         <<<NA/4,      128, 0, stream>>>(P);
  k_edge_fwd   <<<NA/4,      128, 0, stream>>>(P);
  k_fit_fwd    <<<NA/128,    128, 0, stream>>>(P);
  k_fit_bwd    <<<NA/128,    128, 0, stream>>>(P);
  k_grho       <<<(BB*NFREQ)/256, 256, 0, stream>>>(P);
  k_nufft_force<<<NA/4,      128, 0, stream>>>(P);
  k_edge_bwd   <<<NA/4,      128, 0, stream>>>(P);
}
